// SelfAttentionLayer_17188459119357
// MI455X (gfx1250) — matrix operand order repaired and checked by reading, not compiled, not run on hardware
//
#include <hip/hip_runtime.h>
#include <stdint.h>

typedef __attribute__((ext_vector_type(16))) __bf16 bf16x16;
typedef __attribute__((ext_vector_type(8)))  __bf16 bf16x8;
typedef __attribute__((ext_vector_type(8)))  float  f32x8;
typedef __attribute__((ext_vector_type(4)))  unsigned int u32x4;
typedef __attribute__((ext_vector_type(8)))  int i32x8;
typedef __attribute__((ext_vector_type(4)))  int i32x4;

#define N_TOK   4096
#define DMODEL  1024
#define NHEADS  16
#define DKV     64
#define KB_KEYS 64   // keys per staged KV tile

__device__ __forceinline__ f32x8 zero8() {
  f32x8 z;
#pragma unroll
  for (int i = 0; i < 8; ++i) z[i] = 0.0f;
  return z;
}

__device__ __forceinline__ f32x8 wmma_bf16(bf16x16 a, bf16x16 b, f32x8 c) {
  return __builtin_amdgcn_wmma_f32_16x16x32_bf16(
      false, a, false, b, (short)0, c, false, false);
}

// 16x32 A tile from row-major storage (ld elements/row): two 16B chunks/lane.
__device__ __forceinline__ bf16x16 load_A(const __bf16* src, int ld) {
  const int lane = threadIdx.x & 31;
  const int m = lane & 15, hf = lane >> 4;
  const __bf16* p = src + (long long)m * ld + hf * 8;
  bf16x8 lo = *(const bf16x8*)(p);
  bf16x8 hi = *(const bf16x8*)(p + 16);
  bf16x16 r;
#pragma unroll
  for (int i = 0; i < 8; ++i) { r[i] = lo[i]; r[i + 8] = hi[i]; }
  return r;
}

// 32x16 B tile from B^T row-major [N][K] storage: two 16B chunks/lane.
__device__ __forceinline__ bf16x16 load_B(const __bf16* srcT, int ld) {
  const int lane = threadIdx.x & 31;
  const int n = lane & 15, hf = lane >> 4;
  const __bf16* p = srcT + (long long)n * ld + hf * 8;
  bf16x8 lo = *(const bf16x8*)(p);
  bf16x8 hi = *(const bf16x8*)(p + 16);
  bf16x16 r;
#pragma unroll
  for (int i = 0; i < 8; ++i) { r[i] = lo[i]; r[i + 8] = hi[i]; }
  return r;
}

// ---- Tensor Data Mover: 2-D tile (d0 contiguous elems per row, d1 rows,
// row stride stride0 elems, 2-byte elements) global -> LDS.
// D# layout per CDNA5 ISA.
__device__ __forceinline__ unsigned lds_off(const void* p) {
  return (unsigned)(unsigned long long)(uintptr_t)p;  // low 32 bits = LDS byte addr
}

__device__ __forceinline__ void tdm_load_2d(unsigned lds_addr, const void* gptr,
                                            unsigned d0, unsigned d1,
                                            unsigned long long stride0) {
  unsigned long long ga = (unsigned long long)(uintptr_t)gptr;
  u32x4 g0;
  g0[0] = 1u;                                   // count=1 (valid user descriptor)
  g0[1] = lds_addr;                             // lds_addr [63:32]
  g0[2] = (unsigned)(ga & 0xffffffffu);         // global_addr[31:0]
  g0[3] = (unsigned)((ga >> 32) & 0x01ffffffu)  // global_addr[56:32]
          | (2u << 30);                         // type=2 ("image")
  i32x8 g1;
  g1[0] = (int)(1u << 16);                      // data_size=1 -> 2 bytes
  g1[1] = (int)((d0 & 0xffffu) << 16);          // tensor_dim0[15:0]
  g1[2] = (int)((d0 >> 16) | ((d1 & 0xffffu) << 16));   // td0[31:16] | td1[15:0]
  g1[3] = (int)((d1 >> 16) | ((d0 & 0xffffu) << 16));   // td1[31:16] | tile_dim0
  g1[4] = (int)(d1 & 0xffffu);                  // tile_dim1 (tile_dim2=0)
  g1[5] = (int)(unsigned)(stride0 & 0xffffffffu);       // tensor_dim0_stride[31:0]
  g1[6] = (int)(unsigned)((stride0 >> 32) & 0xffffu);   // stride[47:32] (dim1_stride=0)
  g1[7] = 0;
  i32x4 z4 = {0, 0, 0, 0};
#if __has_include(<hip/amd_detail/amd_gfx1250_TDM.h>)
  i32x8 z8 = {0, 0, 0, 0, 0, 0, 0, 0};
  __builtin_amdgcn_tensor_load_to_lds(g0, g1, z4, z4, z8, 0);
#else
  __builtin_amdgcn_tensor_load_to_lds(g0, g1, z4, z4, 0);
#endif
}

// ---------------- conversion kernels ----------------

__global__ void cvt_bf16(const float* __restrict__ src, __bf16* __restrict__ dst, int count) {
  int i = blockIdx.x * blockDim.x + threadIdx.x;
  int stride = gridDim.x * blockDim.x;
  for (; i < count; i += stride) dst[i] = (__bf16)src[i];
}

// w[h][D][K] (f32) -> wT[h][K][D] (bf16)
__global__ void cvt_transpose(const float* __restrict__ w, __bf16* __restrict__ wT,
                              int H, int D, int K) {
  int total = H * D * K;
  int i = blockIdx.x * blockDim.x + threadIdx.x;
  int stride = gridDim.x * blockDim.x;
  for (; i < total; i += stride) {
    int h = i / (D * K);
    int r = i - h * D * K;
    int d = r / K;
    int k = r - d * K;
    wT[(long long)h * K * D + (long long)k * D + d] = (__bf16)w[i];
  }
}

// ---------------- QKV projection ----------------
// grid (N/64, H, 3), block 128. Register double-buffered K-loop.
__global__ __launch_bounds__(128) void qkv_proj(
    const __bf16* __restrict__ xb,
    const __bf16* __restrict__ wqT, const __bf16* __restrict__ wkT,
    const __bf16* __restrict__ wvT,
    __bf16* __restrict__ Qg, __bf16* __restrict__ Kg, __bf16* __restrict__ VTg) {
  const int wave = threadIdx.x >> 5;
  const int lane = threadIdx.x & 31;
  const int hf = lane >> 4, ln = lane & 15;
  const int n0 = blockIdx.x * 64 + wave * 16;
  const int hh = blockIdx.y;
  const int which = blockIdx.z;

  const __bf16* wT =
      (which == 0 ? wqT : (which == 1 ? wkT : wvT)) + (long long)hh * DKV * DMODEL;
  const __bf16* xrow = xb + (long long)n0 * DMODEL;

  f32x8 acc0 = zero8(), acc1 = zero8(), acc2 = zero8(), acc3 = zero8();

  bf16x16 a  = load_A(xrow, DMODEL);
  bf16x16 b0 = load_B(wT + 0 * 16 * DMODEL, DMODEL);
  bf16x16 b1 = load_B(wT + 1 * 16 * DMODEL, DMODEL);
  bf16x16 b2 = load_B(wT + 2 * 16 * DMODEL, DMODEL);
  bf16x16 b3 = load_B(wT + 3 * 16 * DMODEL, DMODEL);

  for (int kk = 0; kk < DMODEL - 32; kk += 32) {
    // prefetch next K-slice before consuming current one
    bf16x16 an  = load_A(xrow + kk + 32, DMODEL);
    bf16x16 bn0 = load_B(wT + 0 * 16 * DMODEL + kk + 32, DMODEL);
    bf16x16 bn1 = load_B(wT + 1 * 16 * DMODEL + kk + 32, DMODEL);
    bf16x16 bn2 = load_B(wT + 2 * 16 * DMODEL + kk + 32, DMODEL);
    bf16x16 bn3 = load_B(wT + 3 * 16 * DMODEL + kk + 32, DMODEL);
    acc0 = wmma_bf16(a, b0, acc0);
    acc1 = wmma_bf16(a, b1, acc1);
    acc2 = wmma_bf16(a, b2, acc2);
    acc3 = wmma_bf16(a, b3, acc3);
    a = an; b0 = bn0; b1 = bn1; b2 = bn2; b3 = bn3;
  }
  acc0 = wmma_bf16(a, b0, acc0);
  acc1 = wmma_bf16(a, b1, acc1);
  acc2 = wmma_bf16(a, b2, acc2);
  acc3 = wmma_bf16(a, b3, acc3);

  f32x8 acc[4] = {acc0, acc1, acc2, acc3};
  if (which < 2) {
    __bf16* outp = (which == 0 ? Qg : Kg) + (long long)hh * N_TOK * DKV;
#pragma unroll
    for (int nt = 0; nt < 4; ++nt)
#pragma unroll
      for (int v = 0; v < 8; ++v) {
        int row = n0 + v + 8 * hf;
        outp[(long long)row * DKV + nt * 16 + ln] = (__bf16)acc[nt][v];
      }
  } else {
#pragma unroll
    for (int nt = 0; nt < 4; ++nt)
#pragma unroll
      for (int v = 0; v < 8; ++v) {
        int col = nt * 16 + ln;
        int row = n0 + v + 8 * hf;
        VTg[((long long)hh * DKV + col) * N_TOK + row] = (__bf16)acc[nt][v];
      }
  }
}

// ---------------- flash attention ----------------
// grid (N/64, H), block 128 (4 waves x 16 query rows).
// K / V^T tiles staged into double-buffered LDS by the Tensor Data Mover,
// prefetch overlapped with compute via TENSORcnt.
__global__ __launch_bounds__(128) void flash_attn(
    const __bf16* __restrict__ Qg, const __bf16* __restrict__ Kg,
    const __bf16* __restrict__ VTg, __bf16* __restrict__ Zb) {
  __shared__ __bf16 Kbuf[2][KB_KEYS * DKV];   // [key][d]   8KB each
  __shared__ __bf16 Vbuf[2][DKV * KB_KEYS];   // [dv][key]  8KB each
  __shared__ __bf16 plds[4 * 16 * 32];        // per-wave P transpose tile

  const int wave = threadIdx.x >> 5;
  const int lane = threadIdx.x & 31;
  const int hf = lane >> 4, ln = lane & 15;
  const int hh = blockIdx.y;
  const int row0 = blockIdx.x * 64 + wave * 16;

  const __bf16* Qh = Qg + (long long)hh * N_TOK * DKV;
  const __bf16* Kh = Kg + (long long)hh * N_TOK * DKV;
  const __bf16* Vh = VTg + (long long)hh * DKV * N_TOK;
  __bf16* pl = plds + wave * (16 * 32);

  bf16x16 aq0 = load_A(Qh + (long long)row0 * DKV, DKV);
  bf16x16 aq1 = load_A(Qh + (long long)row0 * DKV + 32, DKV);

  float mrun[8], lsum[8];
#pragma unroll
  for (int v = 0; v < 8; ++v) { mrun[v] = -3.0e38f; lsum[v] = 0.0f; }
  f32x8 o[4];
#pragma unroll
  for (int nt = 0; nt < 4; ++nt) o[nt] = zero8();

  const float scale = 0.125f;  // 1/sqrt(64)
  const int n_iter = N_TOK / KB_KEYS;

  if (wave == 0) {
    tdm_load_2d(lds_off(&Kbuf[0][0]), Kh, DKV, KB_KEYS, DKV);          // K tile
    tdm_load_2d(lds_off(&Vbuf[0][0]), Vh, KB_KEYS, DKV, N_TOK);        // V^T tile
  }

  for (int it = 0; it < n_iter; ++it) {
    const int kb = it * KB_KEYS;
    const int cur = it & 1, nxt = cur ^ 1;

    __syncthreads();  // everyone done reading buffers[nxt] from iteration it-1
    if (wave == 0) {
      if (it + 1 < n_iter) {
        const int kb2 = kb + KB_KEYS;
        tdm_load_2d(lds_off(&Kbuf[nxt][0]), Kh + (long long)kb2 * DKV, DKV, KB_KEYS, DKV);
        tdm_load_2d(lds_off(&Vbuf[nxt][0]), Vh + kb2, KB_KEYS, DKV, N_TOK);
        __builtin_amdgcn_s_wait_tensorcnt(2);  // cur tiles complete (in-order)
      } else {
        __builtin_amdgcn_s_wait_tensorcnt(0);
      }
    }
    __syncthreads();  // cur buffers visible to all waves

    const __bf16* Kt = &Kbuf[cur][0];
    const __bf16* Vt = &Vbuf[cur][0];

#pragma unroll
    for (int sub = 0; sub < 2; ++sub) {   // two 32-key subtiles
      const int ko = sub * 32;
      f32x8 s0 = zero8(), s1 = zero8();
      s0 = wmma_bf16(aq0, load_B(Kt + (long long)ko * DKV, DKV), s0);
      s0 = wmma_bf16(aq1, load_B(Kt + (long long)ko * DKV + 32, DKV), s0);
      s1 = wmma_bf16(aq0, load_B(Kt + (long long)(ko + 16) * DKV, DKV), s1);
      s1 = wmma_bf16(aq1, load_B(Kt + (long long)(ko + 16) * DKV + 32, DKV), s1);

#pragma unroll
      for (int v = 0; v < 8; ++v) {
        float a0 = s0[v] * scale, a1 = s1[v] * scale;
        float t = fmaxf(a0, a1);
#pragma unroll
        for (int off = 8; off >= 1; off >>= 1) t = fmaxf(t, __shfl_xor(t, off, 16));
        float mn = fmaxf(mrun[v], t);
        float alpha = __expf(mrun[v] - mn);
        float p0 = __expf(a0 - mn), p1 = __expf(a1 - mn);
        float rs = p0 + p1;
#pragma unroll
        for (int off = 8; off >= 1; off >>= 1) rs += __shfl_xor(rs, off, 16);
        lsum[v] = lsum[v] * alpha + rs;
        mrun[v] = mn;
#pragma unroll
        for (int nt = 0; nt < 4; ++nt) o[nt][v] *= alpha;
        int row = v + 8 * hf;
        pl[row * 32 + ln] = (__bf16)p0;
        pl[row * 32 + 16 + ln] = (__bf16)p1;
      }

      asm volatile("s_wait_dscnt 0" ::: "memory");
      bf16x16 ap = load_A(pl, 32);  // C-layout -> A-layout via wave-private LDS

#pragma unroll
      for (int nt = 0; nt < 4; ++nt)
        o[nt] = wmma_bf16(ap, load_B(Vt + nt * 16 * KB_KEYS + ko, KB_KEYS), o[nt]);
    }
  }

#pragma unroll
  for (int nt = 0; nt < 4; ++nt)
#pragma unroll
    for (int v = 0; v < 8; ++v) {
      int row = row0 + v + 8 * hf;
      int col = hh * DKV + nt * 16 + ln;
      Zb[(long long)row * DMODEL + col] = (__bf16)(o[nt][v] / lsum[v]);
    }
}

// ---------------- output projection ----------------
// grid (N/64, DMODEL/64), block 128. Register double-buffered K-loop.
__global__ __launch_bounds__(128) void out_proj(
    const __bf16* __restrict__ Zb, const __bf16* __restrict__ woT,
    float* __restrict__ out) {
  const int wave = threadIdx.x >> 5;
  const int lane = threadIdx.x & 31;
  const int hf = lane >> 4, ln = lane & 15;
  const int row0 = blockIdx.x * 64 + wave * 16;
  const int col0 = blockIdx.y * 64;

  const __bf16* zrow = Zb + (long long)row0 * DMODEL;
  const __bf16* w0 = woT + (long long)(col0 + 0) * DMODEL;
  const __bf16* w1 = woT + (long long)(col0 + 16) * DMODEL;
  const __bf16* w2 = woT + (long long)(col0 + 32) * DMODEL;
  const __bf16* w3 = woT + (long long)(col0 + 48) * DMODEL;

  f32x8 acc0 = zero8(), acc1 = zero8(), acc2 = zero8(), acc3 = zero8();

  bf16x16 a  = load_A(zrow, DMODEL);
  bf16x16 b0 = load_B(w0, DMODEL);
  bf16x16 b1 = load_B(w1, DMODEL);
  bf16x16 b2 = load_B(w2, DMODEL);
  bf16x16 b3 = load_B(w3, DMODEL);

  for (int kk = 0; kk < DMODEL - 32; kk += 32) {
    bf16x16 an  = load_A(zrow + kk + 32, DMODEL);
    bf16x16 bn0 = load_B(w0 + kk + 32, DMODEL);
    bf16x16 bn1 = load_B(w1 + kk + 32, DMODEL);
    bf16x16 bn2 = load_B(w2 + kk + 32, DMODEL);
    bf16x16 bn3 = load_B(w3 + kk + 32, DMODEL);
    acc0 = wmma_bf16(a, b0, acc0);
    acc1 = wmma_bf16(a, b1, acc1);
    acc2 = wmma_bf16(a, b2, acc2);
    acc3 = wmma_bf16(a, b3, acc3);
    a = an; b0 = bn0; b1 = bn1; b2 = bn2; b3 = bn3;
  }
  acc0 = wmma_bf16(a, b0, acc0);
  acc1 = wmma_bf16(a, b1, acc1);
  acc2 = wmma_bf16(a, b2, acc2);
  acc3 = wmma_bf16(a, b3, acc3);

  f32x8 acc[4] = {acc0, acc1, acc2, acc3};
#pragma unroll
  for (int nt = 0; nt < 4; ++nt)
#pragma unroll
    for (int v = 0; v < 8; ++v) {
      int row = row0 + v + 8 * hf;
      out[(long long)row * DMODEL + col0 + nt * 16 + ln] = acc[nt][v];
    }
}

// ---------------- launch ----------------

extern "C" void kernel_launch(void* const* d_in, const int* in_sizes, int n_in,
                              void* d_out, int out_size, void* d_ws, size_t ws_size,
                              hipStream_t stream) {
  const float* x  = (const float*)d_in[0];
  const float* wq = (const float*)d_in[1];
  const float* wk = (const float*)d_in[2];
  const float* wv = (const float*)d_in[3];
  const float* wo = (const float*)d_in[4];
  float* out = (float*)d_out;

  char* ws = (char*)d_ws;
  size_t off = 0;
  auto alloc = [&](size_t bytes) -> void* {
    void* p = ws + off;
    off += (bytes + 255) & ~(size_t)255;
    return p;
  };

  __bf16* xb  = (__bf16*)alloc((size_t)N_TOK * DMODEL * 2);
  __bf16* wqT = (__bf16*)alloc((size_t)NHEADS * DKV * DMODEL * 2);
  __bf16* wkT = (__bf16*)alloc((size_t)NHEADS * DKV * DMODEL * 2);
  __bf16* wvT = (__bf16*)alloc((size_t)NHEADS * DKV * DMODEL * 2);
  __bf16* woT = (__bf16*)alloc((size_t)DMODEL * DMODEL * 2);
  __bf16* Qg  = (__bf16*)alloc((size_t)NHEADS * N_TOK * DKV * 2);
  __bf16* Kg  = (__bf16*)alloc((size_t)NHEADS * N_TOK * DKV * 2);
  __bf16* VTg = (__bf16*)alloc((size_t)NHEADS * DKV * N_TOK * 2);
  __bf16* Zb  = (__bf16*)alloc((size_t)N_TOK * DMODEL * 2);
  (void)ws_size; (void)in_sizes; (void)n_in; (void)out_size;

  cvt_bf16<<<2048, 256, 0, stream>>>(x, xb, N_TOK * DMODEL);
  cvt_transpose<<<1024, 256, 0, stream>>>(wq, wqT, NHEADS, DMODEL, DKV);
  cvt_transpose<<<1024, 256, 0, stream>>>(wk, wkT, NHEADS, DMODEL, DKV);
  cvt_transpose<<<1024, 256, 0, stream>>>(wv, wvT, NHEADS, DMODEL, DKV);
  cvt_transpose<<<1024, 256, 0, stream>>>(wo, woT, 1, DMODEL, DMODEL);

  qkv_proj<<<dim3(N_TOK / 64, NHEADS, 3), 128, 0, stream>>>(xb, wqT, wkT, wvT, Qg, Kg, VTg);
  flash_attn<<<dim3(N_TOK / 64, NHEADS), 128, 0, stream>>>(Qg, Kg, VTg, Zb);
  out_proj<<<dim3(N_TOK / 64, DMODEL / 64), 128, 0, stream>>>(Zb, woT, out);
}
